// CNN_Front_Attention_Mean_Universal_67585605369987
// MI455X (gfx1250) — compile-verified, matrix operand order checked by reading, not run on hardware
//
#include <hip/hip_runtime.h>
#include <hip/hip_bf16.h>

typedef __bf16 bf16;
typedef __attribute__((ext_vector_type(16))) __bf16 v16bf;
typedef __attribute__((ext_vector_type(8)))  __bf16 v8bf;
typedef __attribute__((ext_vector_type(8)))  float  v8f;

#define WMMA_BF16(a, b, c) \
  __builtin_amdgcn_wmma_f32_16x16x32_bf16(false, (a), false, (b), (short)0, (c), false, false)

// ---------------------------------------------------------------------------
// Fragment load: per CDNA5 ISA 7.12.2, a 16-bit A/B fragment for a lane is two
// contiguous 16-byte runs (K = kh..kh+7 and kh+16..kh+23, kh = (lane>=16)*8).
// Source data is staged in LDS so that the lane's K-run is contiguous ->
// two ds_load_b128 per fragment.
// ---------------------------------------------------------------------------
__device__ __forceinline__ v16bf frag_ld(const bf16* base) {
  v8bf lo = *(const v8bf*)(base);
  v8bf hi = *(const v8bf*)(base + 16);
  v16bf r;
#pragma unroll
  for (int j = 0; j < 8; ++j) { r[j] = lo[j]; r[8 + j] = hi[j]; }
  return r;
}

// A-style fragment: lane holds row (lane&15), K contiguous at row*ld + kh.
__device__ __forceinline__ v16bf frag_row(const bf16* s, int row0, int ld, int koff) {
  int lane = threadIdx.x & 31;
  int row  = row0 + (lane & 15);
  int kh   = (lane >> 4) << 3;
  return frag_ld(s + row * ld + koff + kh);
}

// ---------------------------------------------------------------------------
// f32 -> bf16 conversion (plain) and transposing weight conversion.
// dst layout for weights: [L][N][K]  (B-operand friendly)
// ---------------------------------------------------------------------------
__global__ void k_f32_to_bf16(const float* __restrict__ src, bf16* __restrict__ dst, int n) {
  int i = blockIdx.x * 256 + threadIdx.x;
  if (i < n) dst[i] = (bf16)src[i];
}

__global__ void k_wtrans(const float* __restrict__ src, bf16* __restrict__ dst,
                         int L, int K, int N) {
  int idx = blockIdx.x * 256 + threadIdx.x;
  if (idx >= L * K * N) return;
  int li = idx / (K * N), r = idx % (K * N);
  int n = r / K, k = r % K;                       // write-contiguous along K
  dst[idx] = (bf16)src[(size_t)li * K * N + (size_t)k * N + n];
}

// ---------------------------------------------------------------------------
// Complex LayerNorm (whitening). One wave per row of F=256. Outputs bf16.
// ---------------------------------------------------------------------------
__global__ void k_cln(const float* __restrict__ Xr, const float* __restrict__ Xi,
                      bf16* __restrict__ Nr, bf16* __restrict__ Ni) {
  int wave = threadIdx.x >> 5;
  int lane = threadIdx.x & 31;
  int row  = blockIdx.x * 8 + wave;
  const float* xr = Xr + (size_t)row * 256 + lane * 8;
  const float* xi = Xi + (size_t)row * 256 + lane * 8;
  float vr[8], vi[8];
  float4 a0 = *(const float4*)(xr), a1 = *(const float4*)(xr + 4);
  float4 b0 = *(const float4*)(xi), b1 = *(const float4*)(xi + 4);
  vr[0]=a0.x; vr[1]=a0.y; vr[2]=a0.z; vr[3]=a0.w; vr[4]=a1.x; vr[5]=a1.y; vr[6]=a1.z; vr[7]=a1.w;
  vi[0]=b0.x; vi[1]=b0.y; vi[2]=b0.z; vi[3]=b0.w; vi[4]=b1.x; vi[5]=b1.y; vi[6]=b1.z; vi[7]=b1.w;
  float sr=0, si=0, srr=0, sii=0, sri=0;
#pragma unroll
  for (int j = 0; j < 8; ++j) {
    sr += vr[j]; si += vi[j];
    srr += vr[j]*vr[j]; sii += vi[j]*vi[j]; sri += vr[j]*vi[j];
  }
#pragma unroll
  for (int m = 1; m <= 16; m <<= 1) {
    sr  += __shfl_xor(sr,  m, 32); si  += __shfl_xor(si,  m, 32);
    srr += __shfl_xor(srr, m, 32); sii += __shfl_xor(sii, m, 32);
    sri += __shfl_xor(sri, m, 32);
  }
  const float inv256 = 1.0f / 256.0f;
  float mr = sr * inv256, mi = si * inv256;
  float Vrr = srr * inv256 - mr * mr + 1e-5f;
  float Vii = sii * inv256 - mi * mi + 1e-5f;
  float Vri = sri * inv256 - mr * mi;
  float s  = sqrtf(fmaxf(Vrr * Vii - Vri * Vri, 1e-20f));
  float t  = sqrtf(Vrr + Vii + 2.0f * s);
  float inv = 1.0f / (s * t);
  float Wrr = (Vii + s) * inv, Wii = (Vrr + s) * inv, Wri = -Vri * inv;
  bf16* nr = Nr + (size_t)row * 256 + lane * 8;
  bf16* ni = Ni + (size_t)row * 256 + lane * 8;
#pragma unroll
  for (int j = 0; j < 8; ++j) {
    float cr = vr[j] - mr, ci = vi[j] - mi;
    nr[j] = (bf16)(Wrr * cr + Wri * ci);
    ni[j] = (bf16)(Wri * cr + Wii * ci);
  }
}

// ---------------------------------------------------------------------------
// Generic complex GEMM: Yr = Xr Wr - Xi Wi + br ; Yi = Xr Wi + Xi Wr + bi
// X: bf16 [M][K] row-major.  W: bf16 TRANSPOSED [N][K].
// Block 256 thr = 8 waves, 64x64 tile, f32 WMMA accumulation.
// transpose_out: 0 = identity, 1 = (b,c,t)->(b,t,c) row remap,
//                2 = write V transposed for attention: [(bc,h,d)][t]
// ---------------------------------------------------------------------------
__device__ __forceinline__ int tmap_bct_btc(int m) {  // (b*8+c)*256+t -> (b*256+t)*8+c
  int b = m >> 11, c = (m >> 8) & 7, t = m & 255;
  return ((b << 8) + t) * 8 + c;
}

__global__ void k_cgemm(int M, int N, int K,
                        const bf16* __restrict__ Xr, const bf16* __restrict__ Xi,
                        const bf16* __restrict__ Wr, const bf16* __restrict__ Wi,
                        const float* __restrict__ br, const float* __restrict__ bi,
                        float* OfR, float* OfI, bf16* ObR, bf16* ObI,
                        const float* Rr, const float* Ri,
                        int transpose_out, int leaky) {
  __shared__ __attribute__((aligned(16))) bf16 sXr[64 * 32], sXi[64 * 32];
  __shared__ __attribute__((aligned(16))) bf16 sWr[64 * 32], sWi[64 * 32];  // [n][k]
  int tid = threadIdx.x;
  int bn = blockIdx.x, bm = blockIdx.y;
  int w = tid >> 5, wm = w & 3, wn = w >> 2;     // 4 M-waves x 2 N-waves
  v8f accRR[2] = {}, accII[2] = {}, accRI[2] = {}, accIR[2] = {};

  int lr = tid >> 2, lseg = (tid & 3) << 3;      // 64 rows x 4 x 8 halves
  for (int kt = 0; kt < K / 32; ++kt) {
    __syncthreads();
    {
      size_t g = (size_t)(bm * 64 + lr) * K + kt * 32 + lseg;
      *(uint4*)&sXr[lr * 32 + lseg] = *(const uint4*)(Xr + g);
      *(uint4*)&sXi[lr * 32 + lseg] = *(const uint4*)(Xi + g);
    }
    {
      size_t g = (size_t)(bn * 64 + lr) * K + kt * 32 + lseg;   // W is [N][K]
      *(uint4*)&sWr[lr * 32 + lseg] = *(const uint4*)(Wr + g);
      *(uint4*)&sWi[lr * 32 + lseg] = *(const uint4*)(Wi + g);
    }
    __syncthreads();
    v16bf axr = frag_row(sXr, wm * 16, 32, 0);
    v16bf axi = frag_row(sXi, wm * 16, 32, 0);
#pragma unroll
    for (int j = 0; j < 2; ++j) {
      v16bf bwr = frag_row(sWr, wn * 32 + j * 16, 32, 0);
      v16bf bwi = frag_row(sWi, wn * 32 + j * 16, 32, 0);
      accRR[j] = WMMA_BF16(axr, bwr, accRR[j]);
      accII[j] = WMMA_BF16(axi, bwi, accII[j]);
      accRI[j] = WMMA_BF16(axr, bwi, accRI[j]);
      accIR[j] = WMMA_BF16(axi, bwr, accIR[j]);
    }
  }

  int lane = tid & 31, half = lane >> 4, nn = lane & 15;
#pragma unroll
  for (int j = 0; j < 2; ++j)
#pragma unroll
    for (int g = 0; g < 8; ++g) {
      int m = bm * 64 + wm * 16 + g + 8 * half;
      int n = bn * 64 + wn * 32 + j * 16 + nn;
      float yr = accRR[j][g] - accII[j][g] + br[n];
      float yi = accRI[j][g] + accIR[j][g] + bi[n];
      if (leaky) {
        yr = yr > 0.f ? yr : 0.01f * yr;
        yi = yi > 0.f ? yi : 0.01f * yi;
      }
      if (Rr) { yr += Rr[(size_t)m * N + n]; yi += Ri[(size_t)m * N + n]; }
      size_t o;
      if (transpose_out == 1) {
        o = (size_t)tmap_bct_btc(m) * N + n;
      } else if (transpose_out == 2) {           // V for time attention
        int bc = m >> 8, t = m & 255, h = n >> 6, d = n & 63;
        o = ((size_t)(bc * 4 + h) * 64 + d) * 256 + t;
      } else {
        o = (size_t)m * N + n;
      }
      if (OfR) { OfR[o] = yr; OfI[o] = yi; }
      if (ObR) { ObR[o] = (bf16)yr; ObI[o] = (bf16)yi; }
    }
}

// ---------------------------------------------------------------------------
// Fused complex flash attention over time: T=256, dk=64, per-head.
// grid = (T/64, H=4, B*C=64), block = 128 (4 waves x 16 query rows).
// Q,K in [bc*T + t][f] layout; V pre-transposed: [(bc*4+h)*64 + d][t].
// ---------------------------------------------------------------------------
__global__ void k_attn_time(const bf16* __restrict__ Qr, const bf16* __restrict__ Qi,
                            const bf16* __restrict__ Kr, const bf16* __restrict__ Ki,
                            const bf16* __restrict__ Vr, const bf16* __restrict__ Vi,
                            bf16* __restrict__ Or, bf16* __restrict__ Oi) {
  __shared__ __attribute__((aligned(16))) bf16 sqr[64 * 64], sqi[64 * 64];
  __shared__ __attribute__((aligned(16))) bf16 skr[32 * 64], ski[32 * 64];  // [key][dk]
  __shared__ __attribute__((aligned(16))) bf16 svr[64 * 32], svi[64 * 32];  // [d][key]
  __shared__ __attribute__((aligned(16))) bf16 spR[4][16 * 32], spI[4][16 * 32];

  int tid = threadIdx.x;
  int q0 = blockIdx.x * 64;
  int hcol = blockIdx.y * 64;
  int bcBase = blockIdx.z * 256;                                  // bc * T
  size_t vbase = ((size_t)(blockIdx.z * 4 + blockIdx.y) * 64) * 256;
  int w = tid >> 5, lane = tid & 31, half = lane >> 4, nn = lane & 15;
  bf16* prw = &spR[w][0];
  bf16* piw = &spI[w][0];

  // load 64x64 q tile (both r and i)
#pragma unroll
  for (int i = 0; i < 4; ++i) {
    int idx = tid + i * 128;
    int r = idx >> 3, seg = (idx & 7) << 3;
    size_t g = (size_t)(bcBase + q0 + r) * 256 + hcol + seg;
    *(uint4*)&sqr[r * 64 + seg] = *(const uint4*)(Qr + g);
    *(uint4*)&sqi[r * 64 + seg] = *(const uint4*)(Qi + g);
  }
  __syncthreads();
  v16bf aqr0 = frag_row(sqr, w * 16, 64, 0), aqr1 = frag_row(sqr, w * 16, 64, 32);
  v16bf aqi0 = frag_row(sqi, w * 16, 64, 0), aqi1 = frag_row(sqi, w * 16, 64, 32);

  v8f Arr[4] = {}, Aii[4] = {}, Ari[4] = {}, Air[4] = {};
  float mR[8], lR[8], mI[8], lI[8];
#pragma unroll
  for (int g = 0; g < 8; ++g) { mR[g] = -3.0e38f; mI[g] = -3.0e38f; lR[g] = 0.f; lI[g] = 0.f; }

  for (int kc = 0; kc < 8; ++kc) {   // 8 chunks of 32 keys
    __syncthreads();
    {
      // K chunk: 32 rows x 64 dk, row-major
      int idx = tid * 2;                       // 128 thr x 2 uint4 = 256
      int r = idx >> 3, seg = (idx & 7) << 3;
      size_t g = (size_t)(bcBase + kc * 32 + r) * 256 + hcol + seg;
      *(uint4*)&skr[r * 64 + seg]     = *(const uint4*)(Kr + g);
      *(uint4*)&skr[r * 64 + seg + 8] = *(const uint4*)(Kr + g + 8);
      *(uint4*)&ski[r * 64 + seg]     = *(const uint4*)(Ki + g);
      *(uint4*)&ski[r * 64 + seg + 8] = *(const uint4*)(Ki + g + 8);
      // V chunk from transposed layout: 64 d-rows x 32 keys
      int d = tid >> 1, vseg = (tid & 1) << 4;  // 2 x (16 keys) per d-row
      size_t gv = vbase + (size_t)d * 256 + kc * 32 + vseg;
      *(uint4*)&svr[d * 32 + vseg]     = *(const uint4*)(Vr + gv);
      *(uint4*)&svr[d * 32 + vseg + 8] = *(const uint4*)(Vr + gv + 8);
      *(uint4*)&svi[d * 32 + vseg]     = *(const uint4*)(Vi + gv);
      *(uint4*)&svi[d * 32 + vseg + 8] = *(const uint4*)(Vi + gv + 8);
    }
    __syncthreads();

    // scores: sr = (qr kr^T - qi ki^T)*scale ; si = (qr ki^T + qi kr^T)*scale
    float srj[2][8], sij[2][8];
#pragma unroll
    for (int j = 0; j < 2; ++j) {
      v16bf bkr0 = frag_row(skr, j * 16, 64, 0), bkr1 = frag_row(skr, j * 16, 64, 32);
      v16bf bki0 = frag_row(ski, j * 16, 64, 0), bki1 = frag_row(ski, j * 16, 64, 32);
      v8f prr = {}; prr = WMMA_BF16(aqr0, bkr0, prr); prr = WMMA_BF16(aqr1, bkr1, prr);
      v8f pii = {}; pii = WMMA_BF16(aqi0, bki0, pii); pii = WMMA_BF16(aqi1, bki1, pii);
      v8f pri = {}; pri = WMMA_BF16(aqr0, bki0, pri); pri = WMMA_BF16(aqr1, bki1, pri);
      v8f pir = {}; pir = WMMA_BF16(aqi0, bkr0, pir); pir = WMMA_BF16(aqi1, bkr1, pir);
#pragma unroll
      for (int g = 0; g < 8; ++g) {
        srj[j][g] = (prr[g] - pii[g]) * 0.125f;
        sij[j][g] = (pri[g] + pir[g]) * 0.125f;
      }
    }

    // dual online softmax (independent real/imag streams)
#pragma unroll
    for (int g = 0; g < 8; ++g) {
      {
        float s0 = srj[0][g], s1 = srj[1][g];
        float lm = fmaxf(s0, s1);
        lm = fmaxf(lm, __shfl_xor(lm, 1, 32)); lm = fmaxf(lm, __shfl_xor(lm, 2, 32));
        lm = fmaxf(lm, __shfl_xor(lm, 4, 32)); lm = fmaxf(lm, __shfl_xor(lm, 8, 32));
        float mn = fmaxf(mR[g], lm);
        float al = __expf(mR[g] - mn);
        float p0 = __expf(s0 - mn), p1 = __expf(s1 - mn);
        float ps = p0 + p1;
        ps += __shfl_xor(ps, 1, 32); ps += __shfl_xor(ps, 2, 32);
        ps += __shfl_xor(ps, 4, 32); ps += __shfl_xor(ps, 8, 32);
        lR[g] = lR[g] * al + ps; mR[g] = mn;
        prw[(g + 8 * half) * 32 + nn]      = (bf16)p0;
        prw[(g + 8 * half) * 32 + 16 + nn] = (bf16)p1;
#pragma unroll
        for (int jb = 0; jb < 4; ++jb) { Arr[jb][g] *= al; Ari[jb][g] *= al; }
      }
      {
        float s0 = sij[0][g], s1 = sij[1][g];
        float lm = fmaxf(s0, s1);
        lm = fmaxf(lm, __shfl_xor(lm, 1, 32)); lm = fmaxf(lm, __shfl_xor(lm, 2, 32));
        lm = fmaxf(lm, __shfl_xor(lm, 4, 32)); lm = fmaxf(lm, __shfl_xor(lm, 8, 32));
        float mn = fmaxf(mI[g], lm);
        float al = __expf(mI[g] - mn);
        float p0 = __expf(s0 - mn), p1 = __expf(s1 - mn);
        float ps = p0 + p1;
        ps += __shfl_xor(ps, 1, 32); ps += __shfl_xor(ps, 2, 32);
        ps += __shfl_xor(ps, 4, 32); ps += __shfl_xor(ps, 8, 32);
        lI[g] = lI[g] * al + ps; mI[g] = mn;
        piw[(g + 8 * half) * 32 + nn]      = (bf16)p0;
        piw[(g + 8 * half) * 32 + 16 + nn] = (bf16)p1;
#pragma unroll
        for (int jb = 0; jb < 4; ++jb) { Aii[jb][g] *= al; Air[jb][g] *= al; }
      }
    }
    __syncthreads();

    // P @ V  (one k=32 WMMA per 16-wide output block)
    v16bf par = frag_row(prw, 0, 32, 0);
    v16bf pai = frag_row(piw, 0, 32, 0);
#pragma unroll
    for (int jb = 0; jb < 4; ++jb) {
      v16bf bvr = frag_row(svr, jb * 16, 32, 0);   // [d][key] -> B frag
      v16bf bvi = frag_row(svi, jb * 16, 32, 0);
      Arr[jb] = WMMA_BF16(par, bvr, Arr[jb]);
      Aii[jb] = WMMA_BF16(pai, bvi, Aii[jb]);
      Ari[jb] = WMMA_BF16(par, bvi, Ari[jb]);
      Air[jb] = WMMA_BF16(pai, bvr, Air[jb]);
    }
  }

  // o_r = Arr/l_r - Aii/l_i ; o_i = Ari/l_r + Air/l_i
#pragma unroll
  for (int jb = 0; jb < 4; ++jb)
#pragma unroll
    for (int g = 0; g < 8; ++g) {
      int m = g + 8 * half;
      float invR = 1.0f / lR[g], invI = 1.0f / lI[g];
      float orv = Arr[jb][g] * invR - Aii[jb][g] * invI;
      float oiv = Ari[jb][g] * invR + Air[jb][g] * invI;
      size_t o = (size_t)(bcBase + q0 + w * 16 + m) * 256 + hcol + jb * 16 + nn;
      Or[o] = (bf16)orv;
      Oi[o] = (bf16)oiv;
    }
}

// ---------------------------------------------------------------------------
// Channel attention: seq = C = 8 (tiny) -> scalar LDS kernel, mask supported.
// grid = B*T blocks, 256 threads (4 heads x 64).
// ---------------------------------------------------------------------------
__global__ void k_attn_chan(const bf16* __restrict__ Qr, const bf16* __restrict__ Qi,
                            const bf16* __restrict__ Kr, const bf16* __restrict__ Ki,
                            const bf16* __restrict__ Vr, const bf16* __restrict__ Vi,
                            const unsigned char* __restrict__ mask,
                            bf16* __restrict__ Or, bf16* __restrict__ Oi) {
  __shared__ __attribute__((aligned(16))) bf16 sq[2][2048], sk[2][2048], sv[2][2048];
  __shared__ float sa[2][4][64];
  int tid = threadIdx.x;
  int bt = blockIdx.x;
  int b = bt >> 8;
  size_t base = (size_t)bt * 2048;   // 8 rows x 256 cols, contiguous
  int off = tid * 8;
  *(uint4*)&sq[0][off] = *(const uint4*)(Qr + base + off);
  *(uint4*)&sq[1][off] = *(const uint4*)(Qi + base + off);
  *(uint4*)&sk[0][off] = *(const uint4*)(Kr + base + off);
  *(uint4*)&sk[1][off] = *(const uint4*)(Ki + base + off);
  *(uint4*)&sv[0][off] = *(const uint4*)(Vr + base + off);
  *(uint4*)&sv[1][off] = *(const uint4*)(Vi + base + off);
  __syncthreads();

  int h = tid >> 6, p = tid & 63, qc = p >> 3, kc = p & 7;
  const bf16* qr = &sq[0][qc * 256 + h * 64];
  const bf16* qi = &sq[1][qc * 256 + h * 64];
  const bf16* kr = &sk[0][kc * 256 + h * 64];
  const bf16* ki = &sk[1][kc * 256 + h * 64];
  float sr = 0.f, si = 0.f;
#pragma unroll 16
  for (int d = 0; d < 64; ++d) {
    float a = (float)qr[d], bq = (float)qi[d], c = (float)kr[d], e = (float)ki[d];
    sr += a * c - bq * e;
    si += a * e + bq * c;
  }
  sr *= 0.125f; si *= 0.125f;
  bool ok = mask[(b << 6) + (qc << 3) + kc] != 0;
  if (!ok) { sr = -3.0e38f; si = -3.0e38f; }
  float mv, ev, lv, ar, ai;
  mv = sr;
  mv = fmaxf(mv, __shfl_xor(mv, 1, 32)); mv = fmaxf(mv, __shfl_xor(mv, 2, 32));
  mv = fmaxf(mv, __shfl_xor(mv, 4, 32));
  ev = __expf(sr - mv); lv = ev;
  lv += __shfl_xor(lv, 1, 32); lv += __shfl_xor(lv, 2, 32); lv += __shfl_xor(lv, 4, 32);
  ar = ok ? ev / lv : 0.f;
  mv = si;
  mv = fmaxf(mv, __shfl_xor(mv, 1, 32)); mv = fmaxf(mv, __shfl_xor(mv, 2, 32));
  mv = fmaxf(mv, __shfl_xor(mv, 4, 32));
  ev = __expf(si - mv); lv = ev;
  lv += __shfl_xor(lv, 1, 32); lv += __shfl_xor(lv, 2, 32); lv += __shfl_xor(lv, 4, 32);
  ai = ok ? ev / lv : 0.f;
  sa[0][h][(qc << 3) + kc] = ar;
  sa[1][h][(qc << 3) + kc] = ai;
  __syncthreads();

  int oqc = p >> 3, db = (p & 7) << 3;
#pragma unroll
  for (int dd = 0; dd < 8; ++dd) {
    int d = db + dd;
    float orv = 0.f, oiv = 0.f;
#pragma unroll
    for (int k2 = 0; k2 < 8; ++k2) {
      float a_r = sa[0][h][(oqc << 3) + k2], a_i = sa[1][h][(oqc << 3) + k2];
      float v_r = (float)sv[0][k2 * 256 + h * 64 + d];
      float v_i = (float)sv[1][k2 * 256 + h * 64 + d];
      orv += a_r * v_r - a_i * v_i;
      oiv += a_r * v_i + a_i * v_r;
    }
    size_t o = base + (size_t)oqc * 256 + h * 64 + d;
    Or[o] = (bf16)orv;
    Oi[o] = (bf16)oiv;
  }
}

// ---------------------------------------------------------------------------
// Final mean over channel dim; x layout (B,T,C,F), out (2,B,T,F)
// ---------------------------------------------------------------------------
__global__ void k_mean(const float* __restrict__ Xr, const float* __restrict__ Xi,
                       float* __restrict__ out) {
  int idx = blockIdx.x * 256 + threadIdx.x;   // < B*T*F = 524288
  int f = idx & 255, bt = idx >> 8;
  size_t base = (size_t)bt * 2048 + f;
  float sr = 0.f, si = 0.f;
#pragma unroll
  for (int c = 0; c < 8; ++c) { sr += Xr[base + c * 256]; si += Xi[base + c * 256]; }
  out[idx] = sr * 0.125f;
  out[524288 + idx] = si * 0.125f;
}

// ---------------------------------------------------------------------------
// Host-side orchestration
// ---------------------------------------------------------------------------
extern "C" void kernel_launch(void* const* d_in, const int* in_sizes, int n_in,
                              void* d_out, int out_size, void* d_ws, size_t ws_size,
                              hipStream_t stream) {
  const float* x_r   = (const float*)d_in[0];
  const float* x_i   = (const float*)d_in[1];
  const unsigned char* cmask = (const unsigned char*)d_in[2];
  const float* a1Wr  = (const float*)d_in[3];
  const float* a1Wi  = (const float*)d_in[4];
  const float* a1br  = (const float*)d_in[5];
  const float* a1bi  = (const float*)d_in[6];
  const float* a2Wr  = (const float*)d_in[7];
  const float* a2Wi  = (const float*)d_in[8];
  const float* a2br  = (const float*)d_in[9];
  const float* a2bi  = (const float*)d_in[10];
  const float* f1Wr  = (const float*)d_in[11];
  const float* f1Wi  = (const float*)d_in[12];
  const float* f1br  = (const float*)d_in[13];
  const float* f1bi  = (const float*)d_in[14];
  const float* f2Wr  = (const float*)d_in[15];
  const float* f2Wi  = (const float*)d_in[16];
  const float* f2br  = (const float*)d_in[17];
  const float* f2bi  = (const float*)d_in[18];

  const int M = 16384;            // B*C*T rows
  const size_t NE = (size_t)M * 256;

  char* ws = (char*)d_ws;
  size_t off = 0;
  auto alloc = [&](size_t bytes) -> void* {
    void* p = ws + off;
    off += (bytes + 255) & ~(size_t)255;
    return p;
  };

  // weights stored TRANSPOSED: [L][N][K]
  bf16* wA1r = (bf16*)alloc(4 * 65536 * sizeof(bf16));
  bf16* wA1i = (bf16*)alloc(4 * 65536 * sizeof(bf16));
  bf16* wA2r = (bf16*)alloc(4 * 65536 * sizeof(bf16));
  bf16* wA2i = (bf16*)alloc(4 * 65536 * sizeof(bf16));
  bf16* wF1r = (bf16*)alloc(524288 * sizeof(bf16));
  bf16* wF1i = (bf16*)alloc(524288 * sizeof(bf16));
  bf16* wF2r = (bf16*)alloc(524288 * sizeof(bf16));
  bf16* wF2i = (bf16*)alloc(524288 * sizeof(bf16));
  bf16* nr  = (bf16*)alloc(NE * sizeof(bf16));
  bf16* ni  = (bf16*)alloc(NE * sizeof(bf16));
  bf16* qr  = (bf16*)alloc(NE * sizeof(bf16));
  bf16* qi  = (bf16*)alloc(NE * sizeof(bf16));
  bf16* kr  = (bf16*)alloc(NE * sizeof(bf16));
  bf16* ki  = (bf16*)alloc(NE * sizeof(bf16));
  bf16* vr  = (bf16*)alloc(NE * sizeof(bf16));
  bf16* vi  = (bf16*)alloc(NE * sizeof(bf16));
  bf16* aor = (bf16*)alloc(NE * sizeof(bf16));
  bf16* aoi = (bf16*)alloc(NE * sizeof(bf16));
  float* xr_cur = (float*)alloc(NE * sizeof(float));
  float* xi_cur = (float*)alloc(NE * sizeof(float));
  bf16* hr  = (bf16*)alloc((size_t)M * 2048 * sizeof(bf16));
  bf16* hi  = (bf16*)alloc((size_t)M * 2048 * sizeof(bf16));
  (void)ws_size; (void)n_in; (void)in_sizes; (void)out_size;

  // weight conversion to transposed bf16
  k_wtrans<<<1024, 256, 0, stream>>>(a1Wr, wA1r, 4, 256, 256);
  k_wtrans<<<1024, 256, 0, stream>>>(a1Wi, wA1i, 4, 256, 256);
  k_wtrans<<<1024, 256, 0, stream>>>(a2Wr, wA2r, 4, 256, 256);
  k_wtrans<<<1024, 256, 0, stream>>>(a2Wi, wA2i, 4, 256, 256);
  k_wtrans<<<2048, 256, 0, stream>>>(f1Wr, wF1r, 1, 256, 2048);
  k_wtrans<<<2048, 256, 0, stream>>>(f1Wi, wF1i, 1, 256, 2048);
  k_wtrans<<<2048, 256, 0, stream>>>(f2Wr, wF2r, 1, 2048, 256);
  k_wtrans<<<2048, 256, 0, stream>>>(f2Wi, wF2i, 1, 2048, 256);

  dim3 gLin(4, 256);       // N/64, M/64 for N=256
  // ---------------- Block 1: time attention (layout b,c,t) ----------------
  k_cln<<<2048, 256, 0, stream>>>(x_r, x_i, nr, ni);
  bf16* outs_r[3] = {qr, kr, vr};
  bf16* outs_i[3] = {qi, ki, vi};
  for (int li = 0; li < 3; ++li)
    k_cgemm<<<gLin, 256, 0, stream>>>(M, 256, 256, nr, ni,
        wA1r + li * 65536, wA1i + li * 65536, a1br + li * 256, a1bi + li * 256,
        nullptr, nullptr, outs_r[li], outs_i[li], nullptr, nullptr,
        li == 2 ? 2 : 0, 0);                    // V written transposed
  k_attn_time<<<dim3(4, 4, 64), 128, 0, stream>>>(qr, qi, kr, ki, vr, vi, aor, aoi);
  // out-proj + residual, write transposed to (b,t,c) layout
  k_cgemm<<<gLin, 256, 0, stream>>>(M, 256, 256, aor, aoi,
      wA1r + 3 * 65536, wA1i + 3 * 65536, a1br + 768, a1bi + 768,
      xr_cur, xi_cur, nullptr, nullptr, x_r, x_i, 1, 0);

  // ---------------- Block 2: channel attention (layout b,t,c) ------------
  k_cln<<<2048, 256, 0, stream>>>(xr_cur, xi_cur, nr, ni);
  for (int li = 0; li < 3; ++li)
    k_cgemm<<<gLin, 256, 0, stream>>>(M, 256, 256, nr, ni,
        wA2r + li * 65536, wA2i + li * 65536, a2br + li * 256, a2bi + li * 256,
        nullptr, nullptr, outs_r[li], outs_i[li], nullptr, nullptr, 0, 0);
  k_attn_chan<<<2048, 256, 0, stream>>>(qr, qi, kr, ki, vr, vi, cmask, aor, aoi);
  k_cgemm<<<gLin, 256, 0, stream>>>(M, 256, 256, aor, aoi,
      wA2r + 3 * 65536, wA2i + 3 * 65536, a2br + 768, a2bi + 768,
      xr_cur, xi_cur, nullptr, nullptr, xr_cur, xi_cur, 0, 0);

  // ---------------- FFN ---------------------------------------------------
  k_cln<<<2048, 256, 0, stream>>>(xr_cur, xi_cur, nr, ni);
  k_cgemm<<<dim3(32, 256), 256, 0, stream>>>(M, 2048, 256, nr, ni,
      wF1r, wF1i, f1br, f1bi, nullptr, nullptr, hr, hi, nullptr, nullptr, 0, 1);
  k_cgemm<<<gLin, 256, 0, stream>>>(M, 256, 2048, hr, hi,
      wF2r, wF2i, f2br, f2bi, xr_cur, xi_cur, nullptr, nullptr, xr_cur, xi_cur, 0, 0);

  // ---------------- final mean over channels ------------------------------
  k_mean<<<2048, 256, 0, stream>>>(xr_cur, xi_cur, (float*)d_out);
}
